// PopulationLinear_60662118088825
// MI455X (gfx1250) — compile-verified, matrix operand order checked by reading, not run on hardware
//
#include <hip/hip_runtime.h>

#define POP   128
#define D_IN  1024
#define D_OUT 1024
#define BATCH 256

// Block tile: M=128 (batch rows), N=64 (output cols), K-step=32.
#define BM 128
#define BN 64
#define BK 32
#define LDA 40   // padded LDS row stride (bf16 elems): 80B rows break b128 bank conflicts

typedef __attribute__((ext_vector_type(16))) __bf16 v16bf;
typedef __attribute__((ext_vector_type(8)))  float  v8f;

// Pack hi-bf16 halves of two f32 bit patterns into one dword: {hi(f1), hi(f0)}.
// v_perm_b32: S0 supplies byte indices 4-7, S1 supplies 0-3.
__device__ __forceinline__ unsigned int pack_hi2(unsigned int u0, unsigned int u1) {
    return __builtin_amdgcn_perm(u1, u0, 0x07060302u);
}

// Split float4 into packed bf16x2 hi / lo planes (truncating split: f = hi + lo + O(2^-16)).
__device__ __forceinline__ void cvt4(float4 v, uint2& hi, uint2& lo) {
    const unsigned int u0 = __float_as_uint(v.x), u1 = __float_as_uint(v.y);
    const unsigned int u2 = __float_as_uint(v.z), u3 = __float_as_uint(v.w);
    hi.x = pack_hi2(u0, u1);
    hi.y = pack_hi2(u2, u3);
    const float l0 = v.x - __uint_as_float(u0 & 0xffff0000u);   // -> v_fma_mix / v_sub
    const float l1 = v.y - __uint_as_float(u1 & 0xffff0000u);
    const float l2 = v.z - __uint_as_float(u2 & 0xffff0000u);
    const float l3 = v.w - __uint_as_float(u3 & 0xffff0000u);
    lo.x = pack_hi2(__float_as_uint(l0), __float_as_uint(l1));
    lo.y = pack_hi2(__float_as_uint(l2), __float_as_uint(l3));
}

// Load one 16x32 bf16 fragment from LDS following the CDNA5 16-bit A/B layout:
// lanes 0-15: row (lane&15), K = 0..7 (v0-3) and 16..23 (v4-7);
// lanes 16-31: K = 8..15 and 24..31.  Two ds_load_b128 per lane.
__device__ __forceinline__ v16bf lds_frag(const unsigned short* plane_rowbase, int lane) {
    const int r    = lane & 15;
    const int half = lane >> 4;
    const unsigned short* rp = plane_rowbase + r * LDA;
    union { uint4 u[2]; v16bf v; } U;
    U.u[0] = *(const uint4*)(rp + half * 8);
    U.u[1] = *(const uint4*)(rp + 16 + half * 8);
    return U.v;
}

__global__ __launch_bounds__(256)
void poplin_wmma_bf16x3_kernel(const float* __restrict__ x,
                               const float* __restrict__ w,
                               const float* __restrict__ bias,
                               float* __restrict__ out) {
    __shared__ __align__(16) unsigned short As_hi[2][BM * LDA];
    __shared__ __align__(16) unsigned short As_lo[2][BM * LDA];
    __shared__ __align__(16) unsigned short Ws_hi[2][BN * LDA];
    __shared__ __align__(16) unsigned short Ws_lo[2][BN * LDA];

    const int tid  = threadIdx.x;
    const int lane = tid & 31;
    const int wave = tid >> 5;       // 8 waves
    const int mw   = wave & 3;       // wave M tile: mw*32
    const int nw   = wave >> 2;      // wave N tile: nw*32

    const int p  = blockIdx.z;
    const int bm = blockIdx.y * BM;
    const int n0 = blockIdx.x * BN;

    const size_t xstride = (size_t)POP * D_IN;                    // stride between batch rows
    const float* xbase = x + (size_t)bm * xstride + (size_t)p * D_IN;
    const float* wbase = w + ((size_t)p * D_OUT + n0) * D_IN;

    // Precompute per-thread load pointers / LDS store offsets (fixed across K loop).
    const float* xp[4]; int sA[4];
    const float* wp[2]; int sW[2];
#pragma unroll
    for (int j = 0; j < 4; ++j) {
        const int idx = tid + j * 256;
        const int r = idx >> 3, c4 = idx & 7;
        xp[j] = xbase + (size_t)r * xstride + c4 * 4;
        sA[j] = r * LDA + c4 * 4;
    }
#pragma unroll
    for (int j = 0; j < 2; ++j) {
        const int idx = tid + j * 256;
        const int r = idx >> 3, c4 = idx & 7;
        wp[j] = wbase + (size_t)r * D_IN + c4 * 4;
        sW[j] = r * LDA + c4 * 4;
    }

    float4 xr[4], wr[2];

    auto gload = [&](int kt) {
        const int kofs = kt * BK;
#pragma unroll
        for (int j = 0; j < 4; ++j) xr[j] = *(const float4*)(xp[j] + kofs);
#pragma unroll
        for (int j = 0; j < 2; ++j) wr[j] = *(const float4*)(wp[j] + kofs);
        if (kt + 2 < D_IN / BK) {   // pull tile after next into cache -> global_prefetch_b8
            const int kp = (kt + 2) * BK;
            __builtin_prefetch(xp[0] + kp, 0, 3);
            __builtin_prefetch(wp[0] + kp, 0, 3);
        }
    };

    auto cvt_store = [&](int s) {
#pragma unroll
        for (int j = 0; j < 4; ++j) {
            uint2 h, l;
            cvt4(xr[j], h, l);
            *(uint2*)&As_hi[s][sA[j]] = h;
            *(uint2*)&As_lo[s][sA[j]] = l;
        }
#pragma unroll
        for (int j = 0; j < 2; ++j) {
            uint2 h, l;
            cvt4(wr[j], h, l);
            *(uint2*)&Ws_hi[s][sW[j]] = h;
            *(uint2*)&Ws_lo[s][sW[j]] = l;
        }
    };

    v8f acc[2][2];
#pragma unroll
    for (int mi = 0; mi < 2; ++mi)
#pragma unroll
        for (int ni = 0; ni < 2; ++ni)
            acc[mi][ni] = (v8f){0.f, 0.f, 0.f, 0.f, 0.f, 0.f, 0.f, 0.f};

    const int KT = D_IN / BK;   // 32 K-tiles

    // Pipeline prologue: stage 0 resident in LDS, tile 1 in flight in registers.
    gload(0);
    cvt_store(0);
    gload(1);
    __syncthreads();

    for (int kt = 0; kt < KT; ++kt) {
        const int s = kt & 1;

        // ---- compute on stage s (bf16x3: hi*hi + hi*lo + lo*hi) ----
        v16bf ah[2], al[2], bh[2], bl[2];
#pragma unroll
        for (int mi = 0; mi < 2; ++mi) {
            const int rb = (mw * 32 + mi * 16) * LDA;
            ah[mi] = lds_frag(&As_hi[s][rb], lane);
            al[mi] = lds_frag(&As_lo[s][rb], lane);
        }
#pragma unroll
        for (int ni = 0; ni < 2; ++ni) {
            const int rb = (nw * 32 + ni * 16) * LDA;
            bh[ni] = lds_frag(&Ws_hi[s][rb], lane);
            bl[ni] = lds_frag(&Ws_lo[s][rb], lane);
        }
#pragma unroll
        for (int mi = 0; mi < 2; ++mi) {
#pragma unroll
            for (int ni = 0; ni < 2; ++ni) {
                acc[mi][ni] = __builtin_amdgcn_wmma_f32_16x16x32_bf16(
                    false, ah[mi], false, bh[ni], (short)0, acc[mi][ni], false, false);
                acc[mi][ni] = __builtin_amdgcn_wmma_f32_16x16x32_bf16(
                    false, ah[mi], false, bl[ni], (short)0, acc[mi][ni], false, false);
                acc[mi][ni] = __builtin_amdgcn_wmma_f32_16x16x32_bf16(
                    false, al[mi], false, bh[ni], (short)0, acc[mi][ni], false, false);
            }
        }

        // ---- pipeline: commit tile kt+1 to LDS, then launch loads for kt+2 ----
        if (kt + 1 < KT) {
            cvt_store((kt + 1) & 1);
            if (kt + 2 < KT) gload(kt + 2);
        }
        __syncthreads();
    }

    // ---- epilogue: bias add + store (C/D layout: VGPR j -> M = j + 8*(lane>>4), N = lane&15) ----
    const int col_in = (lane & 15);
    float bv[2];
#pragma unroll
    for (int ni = 0; ni < 2; ++ni)
        bv[ni] = bias[(size_t)p * D_OUT + n0 + nw * 32 + ni * 16 + col_in];

#pragma unroll
    for (int mi = 0; mi < 2; ++mi) {
        const int rowbase = bm + mw * 32 + mi * 16 + (lane >> 4) * 8;
#pragma unroll
        for (int ni = 0; ni < 2; ++ni) {
            const int col = n0 + nw * 32 + ni * 16 + col_in;
#pragma unroll
            for (int j = 0; j < 8; ++j) {
                const int b = rowbase + j;
                out[((size_t)b * POP + p) * D_OUT + col] = acc[mi][ni][j] + bv[ni];
            }
        }
    }
}

extern "C" void kernel_launch(void* const* d_in, const int* in_sizes, int n_in,
                              void* d_out, int out_size, void* d_ws, size_t ws_size,
                              hipStream_t stream) {
    (void)in_sizes; (void)n_in; (void)d_ws; (void)ws_size; (void)out_size;
    const float* x    = (const float*)d_in[0];
    const float* w    = (const float*)d_in[1];
    const float* bias = (const float*)d_in[2];
    float* out        = (float*)d_out;

    dim3 grid(D_OUT / BN, BATCH / BM, POP);   // (16, 2, 128)
    poplin_wmma_bf16x3_kernel<<<grid, 256, 0, stream>>>(x, w, bias, out);
}
